// DGCNN_57990648431390
// MI455X (gfx1250) — matrix operand order repaired and checked by reading, not compiled, not run on hardware
//
#include <hip/hip_runtime.h>
#include <hip/hip_bf16.h>

// ---------------------------------------------------------------------------
// DGCNN forward for MI455X (gfx1250): bf16 WMMA for KNN-Gram, EdgeConv2, Conv3.
// Round 2: shfl-tree BN reductions in the dominant GEMM epilogue; async
// global->LDS staging (ASYNCcnt) for the conv3 activation tile.
// ---------------------------------------------------------------------------

typedef __attribute__((ext_vector_type(16))) __bf16 v16bf;
typedef __attribute__((ext_vector_type(8)))  float  v8f;

union AB32 { v16bf v; unsigned int u[8]; };

#define NPTS 512
#define NBATCH 64
#define KNN 20
#define NEGS 0.2f

__device__ inline unsigned short f2bf(float f) {
  unsigned int u = __float_as_uint(f);
  unsigned int r = u + 0x7FFFu + ((u >> 16) & 1u);   // round-to-nearest-even
  return (unsigned short)(r >> 16);
}

// order-preserving float <-> uint encoding for atomicMax on floats
__device__ inline unsigned int fenc(float f) {
  unsigned int u = __float_as_uint(f);
  return (u & 0x80000000u) ? ~u : (u | 0x80000000u);
}
__device__ inline float fdec(unsigned int e) {
  unsigned int u = (e & 0x80000000u) ? (e & 0x7FFFFFFFu) : ~e;
  return __uint_as_float(u);
}

// ---------------------------------------------------------------------------
// Stage 1 KNN: C=3, brute force per point with register insertion top-20.
// ---------------------------------------------------------------------------
__global__ __launch_bounds__(128) void knn1_kernel(const float* __restrict__ x,
                                                   int* __restrict__ idx1) {
  int gid = blockIdx.x * 128 + threadIdx.x;       // (b,i), 32768 total
  int b = gid >> 9, i = gid & 511;
  const float* xb = x + (size_t)b * NPTS * 3;
  float xi0 = xb[i * 3 + 0], xi1 = xb[i * 3 + 1], xi2 = xb[i * 3 + 2];
  float best[KNN]; int bidx[KNN];
#pragma unroll
  for (int s = 0; s < KNN; ++s) { best[s] = -3.4e38f; bidx[s] = 0; }
  for (int j = 0; j < NPTS; ++j) {
    float d0 = xb[j * 3 + 0] - xi0;
    float d1 = xb[j * 3 + 1] - xi1;
    float d2 = xb[j * 3 + 2] - xi2;
    float v = -(d0 * d0 + d1 * d1 + d2 * d2);
    int id = j;
#pragma unroll
    for (int s = 0; s < KNN; ++s) {
      if (v > best[s]) {
        float tv = best[s]; int ti = bidx[s];
        best[s] = v; bidx[s] = id; v = tv; id = ti;
      }
    }
  }
  int* op = idx1 + (size_t)gid * KNN;
#pragma unroll
  for (int s = 0; s < KNN; ++s) op[s] = bidx[s];
}

// ---------------------------------------------------------------------------
// EdgeConv1: 6 -> 64, K=6 too skinny for WMMA; VALU with channel blocking.
// Produces pre-BN max-over-k and per-channel sum/sumsq for training-mode BN.
// ---------------------------------------------------------------------------
__global__ __launch_bounds__(128) void conv1_kernel(const float* __restrict__ x,
                                                    const int* __restrict__ idx1,
                                                    const float* __restrict__ w1,
                                                    float* __restrict__ h1max,
                                                    float* __restrict__ sums1,
                                                    float* __restrict__ sumsq1) {
  __shared__ float sW[64 * 6];
  __shared__ float sSum[64], sSq[64];
  int tid = threadIdx.x;
  if (tid < 64) { sSum[tid] = 0.f; sSq[tid] = 0.f; }
  for (int i = tid; i < 384; i += 128) sW[i] = w1[i];
  __syncthreads();

  int gid = blockIdx.x * 128 + tid;               // (b,n)
  const float* xb = x + (size_t)(gid >> 9) * NPTS * 3;
  int n = gid & 511;
  float xi0 = xb[n * 3 + 0], xi1 = xb[n * 3 + 1], xi2 = xb[n * 3 + 2];
  const int* ip = idx1 + (size_t)gid * KNN;

  for (int cb = 0; cb < 4; ++cb) {
    float hmax[16], hs[16], hq[16];
#pragma unroll
    for (int cc = 0; cc < 16; ++cc) { hmax[cc] = -3.4e38f; hs[cc] = 0.f; hq[cc] = 0.f; }
    for (int k = 0; k < KNN; ++k) {
      int nb = ip[k];
      float f0 = xb[nb * 3 + 0] - xi0;
      float f1 = xb[nb * 3 + 1] - xi1;
      float f2 = xb[nb * 3 + 2] - xi2;
#pragma unroll
      for (int cc = 0; cc < 16; ++cc) {
        const float* w = &sW[(cb * 16 + cc) * 6];
        float h = w[0] * f0 + w[1] * f1 + w[2] * f2 + w[3] * xi0 + w[4] * xi1 + w[5] * xi2;
        hmax[cc] = fmaxf(hmax[cc], h);
        hs[cc] += h; hq[cc] += h * h;
      }
    }
    float* hp = h1max + (size_t)gid * 64 + cb * 16;
#pragma unroll
    for (int cc = 0; cc < 16; ++cc) {
      hp[cc] = hmax[cc];
      atomicAdd(&sSum[cb * 16 + cc], hs[cc]);
      atomicAdd(&sSq[cb * 16 + cc], hq[cc]);
    }
  }
  __syncthreads();
  if (tid < 64) { atomicAdd(&sums1[tid], sSum[tid]); atomicAdd(&sumsq1[tid], sSq[tid]); }
}

// ---------------------------------------------------------------------------
// BN stats -> per-channel affine (scale, bias).
// ---------------------------------------------------------------------------
__global__ void bnstat_kernel(const float* __restrict__ sums, const float* __restrict__ sumsq,
                              const float* __restrict__ g, const float* __restrict__ beta,
                              float* __restrict__ scale, float* __restrict__ bias,
                              int C, float inv_count) {
  int c = blockIdx.x * blockDim.x + threadIdx.x;
  if (c >= C) return;
  float m = sums[c] * inv_count;
  float v = sumsq[c] * inv_count - m * m;
  float sc = g[c] * rsqrtf(v + 1e-5f);
  scale[c] = sc;
  bias[c] = beta[c] - m * sc;
}

// ---------------------------------------------------------------------------
// x1 = lrelu(bn(h1max)) : f32 + bf16 copies + squared norms for KNN-2.
// ---------------------------------------------------------------------------
__global__ __launch_bounds__(128) void x1_kernel(const float* __restrict__ h1max,
                                                 const float* __restrict__ scale,
                                                 const float* __restrict__ bias,
                                                 float* __restrict__ x1f,
                                                 unsigned short* __restrict__ x1bf,
                                                 float* __restrict__ xx1) {
  int gid = blockIdx.x * 128 + threadIdx.x;       // (b,n)
  const float* hp = h1max + (size_t)gid * 64;
  float* xp = x1f + (size_t)gid * 64;
  unsigned short* bp = x1bf + (size_t)gid * 64;
  float acc = 0.f;
#pragma unroll
  for (int c = 0; c < 64; ++c) {
    float v = hp[c] * scale[c] + bias[c];
    v = (v > 0.f) ? v : NEGS * v;
    xp[c] = v;
    bp[c] = f2bf(v);
    acc += v * v;
  }
  xx1[gid] = acc;
}

// ---------------------------------------------------------------------------
// Stage 2 KNN: Gram matrix via v_wmma_f32_16x16x32_bf16 (K=64 -> 2 steps),
// one 16-row x 512-col distance stripe per workgroup, top-20 from LDS.
// ---------------------------------------------------------------------------
__global__ __launch_bounds__(128) void knn2_kernel(const unsigned short* __restrict__ x1bf,
                                                   const float* __restrict__ xx1,
                                                   int* __restrict__ idx2) {
  int b = blockIdx.y;
  int rb = blockIdx.x;                             // 16-row block, 0..31
  int tid = threadIdx.x;
  int lane = tid & 31, wave = tid >> 5;            // 4 waves

  __shared__ unsigned short sA[16][64];
  __shared__ float sXX[NPTS];
  __shared__ float sDist[16][NPTS];

  { // stage the 16 "query" rows (bf16) and all squared norms
    const unsigned int* src = (const unsigned int*)(x1bf + ((size_t)b * NPTS + rb * 16) * 64);
    unsigned int* dst = (unsigned int*)&sA[0][0];
    for (int i = tid; i < 512; i += 128) dst[i] = src[i];
    for (int i = tid; i < NPTS; i += 128) sXX[i] = xx1[(size_t)b * NPTS + i];
  }
  __syncthreads();

  // A fragments: 16x32 bf16, contiguous K per lane half (k = 16h + s), 2 K-steps
  AB32 a0, a1;
  {
    int m = lane & 15;
    int base = (lane >> 4) * 8;
#pragma unroll
    for (int j = 0; j < 8; ++j) {
      int k = 2 * base + 2 * j;
      a0.u[j] = *(const unsigned int*)&sA[m][k];
      a1.u[j] = *(const unsigned int*)&sA[m][k + 32];
    }
  }

  for (int cb = wave; cb < 32; cb += 4) {
    AB32 b0, b1;
    int col = cb * 16 + (lane & 15);
    int ks = (lane >> 4) * 16;
    const unsigned short* bp = x1bf + ((size_t)b * NPTS + col) * 64;
#pragma unroll
    for (int j = 0; j < 8; ++j) {
      b0.u[j] = *(const unsigned int*)&bp[ks + 2 * j];
      b1.u[j] = *(const unsigned int*)&bp[32 + ks + 2 * j];
    }
    v8f c = {};
    c = __builtin_amdgcn_wmma_f32_16x16x32_bf16(false, a0.v, false, b0.v, (short)0, c, false, false);
    c = __builtin_amdgcn_wmma_f32_16x16x32_bf16(false, a1.v, false, b1.v, (short)0, c, false, false);
    int hi = (lane >> 4) * 8;
    int n = cb * 16 + (lane & 15);
#pragma unroll
    for (int r = 0; r < 8; ++r) {
      int m = r + hi;
      sDist[m][n] = 2.0f * c[r] - sXX[rb * 16 + m] - sXX[n];
    }
  }
  __syncthreads();

  if (tid < 16) {
    int row = rb * 16 + tid;
    float best[KNN]; int bidx[KNN];
#pragma unroll
    for (int s = 0; s < KNN; ++s) { best[s] = -3.4e38f; bidx[s] = 0; }
    for (int j = 0; j < NPTS; ++j) {
      float v = sDist[tid][j]; int id = j;
#pragma unroll
      for (int s = 0; s < KNN; ++s) {
        if (v > best[s]) {
          float tv = best[s]; int ti = bidx[s];
          best[s] = v; bidx[s] = id; v = tv; id = ti;
        }
      }
    }
    int* op = idx2 + ((size_t)b * NPTS + row) * KNN;
#pragma unroll
    for (int s = 0; s < KNN; ++s) op[s] = bidx[s];
  }
}

// ---------------------------------------------------------------------------
// Weight f32 -> bf16 conversion (w2: 256x128, w3: 512x320).
// ---------------------------------------------------------------------------
__global__ void wcvt_kernel(const float* __restrict__ w2, const float* __restrict__ w3,
                            unsigned short* __restrict__ w2bf, unsigned short* __restrict__ w3bf) {
  int i = blockIdx.x * blockDim.x + threadIdx.x;
  if (i < 256 * 128) w2bf[i] = f2bf(w2[i]);
  if (i < 512 * 320) w3bf[i] = f2bf(w3[i]);
}

// ---------------------------------------------------------------------------
// EdgeConv2: the dominant GEMM. Per workgroup: 4 points (80 edge columns),
// edge features [128ch x 80col] built in LDS as bf16, W2[256x128] x E.
// 16 M-tiles x 5 N-tiles, 4 K-steps each = 320 WMMAs per workgroup.
// k-max via order-preserving atomicMax; BN sums via shfl trees (lanes 0-15 /
// 16-31 of a C-fragment row share one output channel) -> 16x fewer LDS atomics.
// ---------------------------------------------------------------------------
__global__ __launch_bounds__(256) void conv2_kernel(const float* __restrict__ x1f,
                                                    const int* __restrict__ idx2,
                                                    const unsigned short* __restrict__ w2bf,
                                                    float* __restrict__ h2max,
                                                    float* __restrict__ sums2,
                                                    float* __restrict__ sumsq2) {
  int b = blockIdx.y;
  int n0 = blockIdx.x * 4;
  int tid = threadIdx.x;
  int lane = tid & 31, wave = tid >> 5;            // 8 waves

  __shared__ unsigned short sE[80][128];           // [col][ch] bf16
  __shared__ unsigned int sMax[256 * 4];
  __shared__ float sSum[256], sSq[256];

  if (tid < 256) { sSum[tid] = 0.f; sSq[tid] = 0.f; }
  for (int i = tid; i < 1024; i += 256) sMax[i] = 0u;   // fenc of any finite > 0

  for (int e = tid; e < 80 * 128; e += 256) {
    int col = e >> 7, ch = e & 127;
    int p = col / 20, k = col % 20;
    int nc = n0 + p;
    int nb = idx2[((size_t)b * NPTS + nc) * KNN + k];
    const float* xc = x1f + ((size_t)b * NPTS + nc) * 64;
    const float* xn = x1f + ((size_t)b * NPTS + nb) * 64;
    float v = (ch < 64) ? (xn[ch] - xc[ch]) : xc[ch - 64];
    sE[col][ch] = f2bf(v);
  }
  __syncthreads();

  for (int t = wave; t < 80; t += 8) {             // 10 tiles per wave
    int mt = t / 5, nt = t % 5;
    int mA = mt * 16 + (lane & 15);
    int baseA = (lane >> 4) * 8;
    int colB = nt * 16 + (lane & 15);
    int ksB = (lane >> 4) * 16;
    v8f c = {};
#pragma unroll
    for (int kk = 0; kk < 4; ++kk) {
      AB32 a, bb;
#pragma unroll
      for (int j = 0; j < 8; ++j) {
        int ka = kk * 32 + ksB + 2 * j;
        a.u[j] = *(const unsigned int*)&w2bf[(size_t)mA * 128 + ka];
        int kb = kk * 32 + ksB + 2 * j;
        bb.u[j] = *(const unsigned int*)&sE[colB][kb];
      }
      c = __builtin_amdgcn_wmma_f32_16x16x32_bf16(false, a.v, false, bb.v, (short)0, c, false, false);
    }
    int hi = (lane >> 4) * 8;
    int p = colB / 20;
#pragma unroll
    for (int r = 0; r < 8; ++r) {
      int m = mt * 16 + r + hi;
      float v = c[r];
      atomicMax(&sMax[m * 4 + p], fenc(v));        // k-max needs per-lane resolve
      float s = v, q = v * v;                      // BN sums: 16-lane shfl tree
      s += __shfl_xor(s, 1, 16);  q += __shfl_xor(q, 1, 16);
      s += __shfl_xor(s, 2, 16);  q += __shfl_xor(q, 2, 16);
      s += __shfl_xor(s, 4, 16);  q += __shfl_xor(q, 4, 16);
      s += __shfl_xor(s, 8, 16);  q += __shfl_xor(q, 8, 16);
      if ((lane & 15) == 0) { atomicAdd(&sSum[m], s); atomicAdd(&sSq[m], q); }
    }
  }
  __syncthreads();

  if (tid < 256) {
#pragma unroll
    for (int p = 0; p < 4; ++p)
      h2max[((size_t)b * NPTS + n0 + p) * 256 + tid] = fdec(sMax[tid * 4 + p]);
    atomicAdd(&sums2[tid], sSum[tid]);
    atomicAdd(&sumsq2[tid], sSq[tid]);
  }
}

// ---------------------------------------------------------------------------
// x2 = lrelu(bn(h2max)); build cat = [x1 ; x2] (320 ch) as bf16.
// ---------------------------------------------------------------------------
__global__ __launch_bounds__(128) void x2cat_kernel(const float* __restrict__ h2max,
                                                    const float* __restrict__ scale2,
                                                    const float* __restrict__ bias2,
                                                    const unsigned short* __restrict__ x1bf,
                                                    unsigned short* __restrict__ catbf) {
  int gid = blockIdx.x * 128 + threadIdx.x;        // (b,n)
  const unsigned short* x1p = x1bf + (size_t)gid * 64;
  unsigned short* cp = catbf + (size_t)gid * 320;
#pragma unroll 8
  for (int c = 0; c < 64; ++c) cp[c] = x1p[c];
  const float* hp = h2max + (size_t)gid * 256;
#pragma unroll 8
  for (int c = 0; c < 256; ++c) {
    float v = hp[c] * scale2[c] + bias2[c];
    v = (v > 0.f) ? v : NEGS * v;
    cp[64 + c] = f2bf(v);
  }
}

// ---------------------------------------------------------------------------
// Conv3: W3[512x320] x cat[320x16pts] per workgroup (32 M-tiles, 10 K-steps).
// Activation tile staged via GLOBAL_LOAD_ASYNC_TO_LDS_B128 (ASYNCcnt path);
// BN sums via __shfl_xor lane reductions + global atomics.
// ---------------------------------------------------------------------------
__global__ __launch_bounds__(128) void conv3_kernel(const unsigned short* __restrict__ catbf,
                                                    const unsigned short* __restrict__ w3bf,
                                                    float* __restrict__ h3,
                                                    float* __restrict__ sums3,
                                                    float* __restrict__ sumsq3) {
  int b = blockIdx.y;
  int n0 = blockIdx.x * 16;
  int tid = threadIdx.x;
  int lane = tid & 31, wave = tid >> 5;            // 4 waves

  __shared__ unsigned short sF[16][320];           // [col][ch] bf16, 10240 B
  {
    // async DMA: 128 lanes x 16B x 5 iterations = 10240 B, no VGPR round-trip
    unsigned lds = (unsigned)(size_t)(void*)&sF[0][0] + (unsigned)tid * 16u;
    const char* gp = (const char*)(catbf + ((size_t)b * NPTS + n0) * 320) + tid * 16;
#pragma unroll
    for (int i = 0; i < 5; ++i) {
      asm volatile("global_load_async_to_lds_b128 %0, %1, off"
                   :: "v"(lds + (unsigned)i * 2048u), "v"(gp + i * 2048)
                   : "memory");
    }
    asm volatile("s_wait_asynccnt 0x0" ::: "memory");
  }
  __syncthreads();

  int colB = lane & 15;
  int ksB = (lane >> 4) * 16;
  int baseA = (lane >> 4) * 8;
  for (int mt = wave * 8; mt < wave * 8 + 8; ++mt) {
    int mA = mt * 16 + (lane & 15);
    v8f c = {};
#pragma unroll
    for (int kk = 0; kk < 10; ++kk) {
      AB32 a, bb;
#pragma unroll
      for (int j = 0; j < 8; ++j) {
        int ka = kk * 32 + ksB + 2 * j;
        a.u[j] = *(const unsigned int*)&w3bf[(size_t)mA * 320 + ka];
        int kb = kk * 32 + ksB + 2 * j;
        bb.u[j] = *(const unsigned int*)&sF[colB][kb];
      }
      c = __builtin_amdgcn_wmma_f32_16x16x32_bf16(false, a.v, false, bb.v, (short)0, c, false, false);
    }
#pragma unroll
    for (int r = 0; r < 8; ++r) {
      int m = mt * 16 + r + baseA;
      float v = c[r];
      h3[((size_t)b * NPTS + n0 + colB) * 512 + m] = v;
      float s = v, q = v * v;                       // reduce across the 16 cols
      s += __shfl_xor(s, 1, 16);  q += __shfl_xor(q, 1, 16);
      s += __shfl_xor(s, 2, 16);  q += __shfl_xor(q, 2, 16);
      s += __shfl_xor(s, 4, 16);  q += __shfl_xor(q, 4, 16);
      s += __shfl_xor(s, 8, 16);  q += __shfl_xor(q, 8, 16);
      if ((lane & 15) == 0) { atomicAdd(&sums3[m], s); atomicAdd(&sumsq3[m], q); }
    }
  }
}

// ---------------------------------------------------------------------------
// Pool: out[b, 0:512] = max_n lrelu(bn(h3)), out[b, 512:1024] = mean_n.
// ---------------------------------------------------------------------------
__global__ __launch_bounds__(128) void pool_kernel(const float* __restrict__ h3,
                                                   const float* __restrict__ scale3,
                                                   const float* __restrict__ bias3,
                                                   float* __restrict__ out) {
  int gid = blockIdx.x * 128 + threadIdx.x;        // (b,c)
  int b = gid >> 9, c = gid & 511;
  float sc = scale3[c], bi = bias3[c];
  float mx = -3.4e38f, sm = 0.f;
  const float* hp = h3 + (size_t)b * NPTS * 512 + c;
  for (int n = 0; n < NPTS; ++n) {
    float v = hp[(size_t)n * 512] * sc + bi;
    v = (v > 0.f) ? v : NEGS * v;
    mx = fmaxf(mx, v);
    sm += v;
  }
  out[(size_t)b * 1024 + c] = mx;
  out[(size_t)b * 1024 + 512 + c] = sm * (1.0f / 512.0f);
}

// ---------------------------------------------------------------------------
extern "C" void kernel_launch(void* const* d_in, const int* in_sizes, int n_in,
                              void* d_out, int out_size, void* d_ws, size_t ws_size,
                              hipStream_t stream) {
  (void)in_sizes; (void)n_in; (void)out_size; (void)ws_size;
  const float* x  = (const float*)d_in[0];
  const float* w1 = (const float*)d_in[1];
  const float* g1 = (const float*)d_in[2];
  const float* b1 = (const float*)d_in[3];
  const float* w2 = (const float*)d_in[4];
  const float* g2 = (const float*)d_in[5];
  const float* b2 = (const float*)d_in[6];
  const float* w3 = (const float*)d_in[7];
  const float* g3 = (const float*)d_in[8];
  const float* b3 = (const float*)d_in[9];

  char* ws = (char*)d_ws;
  size_t off = 0;
  auto alloc = [&](size_t bytes) -> void* {
    void* p = ws + off;
    off = (off + bytes + 255) & ~(size_t)255;
    return p;
  };

  int*   idx1  = (int*)  alloc((size_t)NBATCH * NPTS * KNN * 4);
  int*   idx2  = (int*)  alloc((size_t)NBATCH * NPTS * KNN * 4);
  float* h1max = (float*)alloc((size_t)NBATCH * NPTS * 64 * 4);
  float* h2max = (float*)alloc((size_t)NBATCH * NPTS * 256 * 4);
  float* h3    = (float*)alloc((size_t)NBATCH * NPTS * 512 * 4);
  float* x1f   = (float*)alloc((size_t)NBATCH * NPTS * 64 * 4);
  unsigned short* x1bf  = (unsigned short*)alloc((size_t)NBATCH * NPTS * 64 * 2);
  float* xx1   = (float*)alloc((size_t)NBATCH * NPTS * 4);
  unsigned short* catbf = (unsigned short*)alloc((size_t)NBATCH * NPTS * 320 * 2);
  unsigned short* w2bfp = (unsigned short*)alloc((size_t)256 * 128 * 2);
  unsigned short* w3bfp = (unsigned short*)alloc((size_t)512 * 320 * 2);
  float* stats = (float*)alloc((size_t)3328 * 4);

  float* sums1 = stats;          float* sumsq1 = stats + 64;
  float* sc1   = stats + 128;    float* bi1    = stats + 192;
  float* sums2 = stats + 256;    float* sumsq2 = sums2 + 256;
  float* sc2   = sums2 + 512;    float* bi2    = sums2 + 768;
  float* sums3 = sums2 + 1024;   float* sumsq3 = sums3 + 512;
  float* sc3   = sums3 + 1024;   float* bi3    = sums3 + 1536;

  hipMemsetAsync(stats, 0, (size_t)3328 * 4, stream);

  wcvt_kernel<<<640, 256, 0, stream>>>(w2, w3, w2bfp, w3bfp);
  knn1_kernel<<<256, 128, 0, stream>>>(x, idx1);
  conv1_kernel<<<256, 128, 0, stream>>>(x, idx1, w1, h1max, sums1, sumsq1);
  bnstat_kernel<<<1, 64, 0, stream>>>(sums1, sumsq1, g1, b1, sc1, bi1, 64, 1.0f / 655360.0f);
  x1_kernel<<<256, 128, 0, stream>>>(h1max, sc1, bi1, x1f, x1bf, xx1);
  knn2_kernel<<<dim3(32, 64), 128, 0, stream>>>(x1bf, xx1, idx2);
  conv2_kernel<<<dim3(128, 64), 256, 0, stream>>>(x1f, idx2, w2bfp, h2max, sums2, sumsq2);
  bnstat_kernel<<<1, 256, 0, stream>>>(sums2, sumsq2, g2, b2, sc2, bi2, 256, 1.0f / 655360.0f);
  x2cat_kernel<<<256, 128, 0, stream>>>(h2max, sc2, bi2, x1bf, catbf);
  conv3_kernel<<<dim3(32, 64), 128, 0, stream>>>(catbf, w3bfp, h3, sums3, sumsq3);
  bnstat_kernel<<<1, 512, 0, stream>>>(sums3, sumsq3, g3, b3, sc3, bi3, 512, 1.0f / 32768.0f);
  pool_kernel<<<256, 128, 0, stream>>>(h3, sc3, bi3, (float*)d_out);
}
